// DecoderRNNEdges_24472723653047
// MI455X (gfx1250) — matrix operand order repaired and checked by reading, not compiled, not run on hardware
//
#include <hip/hip_runtime.h>
#include <hip/hip_bf16.h>
#include <math.h>

#define HID    1024
#define VOCAB  50000
#define TSTEPS 1023          // scan length = STEPS-1
#define NB1    64            // persistent phase-1 blocks
#define WAVES1 (NB1 * 8)     // 512 waves (wave32)
#define CHUNK  2000          // vocab columns per phase-2 block
#define NCHUNK 25            // 50000 / 2000
#define COLT   125           // 16-col tiles per chunk
#define ROWT   32            // step rows per phase-2 block (2 WMMA tiles)

typedef __attribute__((ext_vector_type(16))) __bf16 v16bf;
typedef __attribute__((ext_vector_type(8)))  float  v8f;

union Frag { uint4 q[2]; v16bf v; };

__device__ __forceinline__ float bf2f(unsigned short u) {
  return __uint_as_float(((unsigned int)u) << 16);
}
__device__ __forceinline__ unsigned short f2bf(float f) {
  unsigned int x = __float_as_uint(f);
  x += 0x7fffu + ((x >> 16) & 1u);            // round-to-nearest-even
  return (unsigned short)(x >> 16);
}

// ---------------- init: zero the grid-barrier counter -----------------------
__global__ void k_init(int* bar) { if (threadIdx.x == 0) *bar = 0; }

// ---------------- f32 -> bf16 weight conversion (one pass, L2-resident) -----
__global__ __launch_bounds__(256) void k_f32bf16(const float* __restrict__ s,
                                                 unsigned short* __restrict__ d,
                                                 size_t n) {
  size_t i = (size_t)blockIdx.x * 256 + threadIdx.x;
  size_t stride = (size_t)gridDim.x * 256;
  for (; i < n; i += stride) d[i] = f2bf(s[i]);
}

// ---------------- Phase 1: persistent GRU over 1023 steps -------------------
__global__ __launch_bounds__(256) void k_gru(
    const int* __restrict__ inputs, const float* __restrict__ z,
    const float* __restrict__ emb_W, const float* __restrict__ h2e_b,
    const float* __restrict__ b_ih, const float* __restrict__ b_hh,
    const unsigned short* __restrict__ wih, const unsigned short* __restrict__ whh,
    const unsigned short* __restrict__ h2e, unsigned short* __restrict__ Ebf,
    float* __restrict__ gA, float* __restrict__ gB, int* __restrict__ bar)
{
  __shared__ float xs[HID];
  __shared__ float hs[HID];
  const int tid  = threadIdx.x;
  const int lane = tid & 31;
  const int gw   = blockIdx.x * 8 + (tid >> 5);

  for (int i = tid; i < HID; i += 256) hs[i] = z[i];
  __syncthreads();

  for (int t = 0; t < TSTEPS; ++t) {
    const int p = t & 1;
    const int tok = (t == 0) ? 0 : inputs[t];           // step_tokens[t]
    for (int i = tid; i < HID; i += 256) xs[i] = emb_W[(size_t)tok * HID + i];
    __syncthreads();

    // 6144 length-1024 dots: rows <3072 -> gi = W_ih.x ; else gh = W_hh.h
    for (int row = gw; row < 6144; row += WAVES1) {
      const unsigned short* wr; const float* vec; float bias; float* dst;
      if (row < 3072) { wr = wih + (size_t)row * HID; vec = xs; bias = b_ih[row]; dst = gA + p * 3072 + row; }
      else { int r2 = row - 3072; wr = whh + (size_t)r2 * HID; vec = hs; bias = b_hh[r2]; dst = gB + p * 3072 + r2; }
      float acc = 0.f;
      #pragma unroll 8
      for (int i = 0; i < 32; ++i) {
        int k2 = lane + (i << 5);
        acc = fmaf(bf2f(wr[k2]), vec[k2], acc);
      }
      #pragma unroll
      for (int m = 16; m >= 1; m >>= 1) acc += __shfl_xor(acc, m, 32);
      if (lane == 0) *dst = acc + bias;
    }

    // ---- grid barrier #t (monotonic arrival counter) ----
    __syncthreads();
    if (tid == 0) {
      __threadfence();
      atomicAdd(bar, 1);
      const int tgt = NB1 * (t + 1);
      volatile int* vb = bar;
      while (*vb < tgt) __builtin_amdgcn_s_sleep(1);
      __threadfence();
    }
    __syncthreads();

    // every block computes full h_new locally (cheap, avoids 2nd barrier)
    const float* ga = gA + p * 3072;
    const float* gb = gB + p * 3072;
    for (int i = tid; i < HID; i += 256) {
      float rr = 1.f / (1.f + __expf(-(ga[i] + gb[i])));
      float zz = 1.f / (1.f + __expf(-(ga[HID + i] + gb[HID + i])));
      float nn = tanhf(ga[2 * HID + i] + rr * gb[2 * HID + i]);
      hs[i] = (1.f - zz) * nn + zz * hs[i];
    }
    __syncthreads();

    // e_t = h2e . h_new + h2e_b -> bf16 row of E
    for (int row = gw; row < HID; row += WAVES1) {
      const unsigned short* wr = h2e + (size_t)row * HID;
      float acc = 0.f;
      #pragma unroll 8
      for (int i = 0; i < 32; ++i) {
        int k2 = lane + (i << 5);
        acc = fmaf(bf2f(wr[k2]), hs[k2], acc);
      }
      #pragma unroll
      for (int m = 16; m >= 1; m >>= 1) acc += __shfl_xor(acc, m, 32);
      if (lane == 0) Ebf[(size_t)t * HID + row] = f2bf(acc + h2e_b[row]);
    }
  }
}

// ---------------- Phase 2: bf16 WMMA GEMM fused w/ online softmax -----------
// M=32 per block: each B fragment feeds two WMMAs. B is pipelined at distance
// 2 (two 1KB fragments in flight over L2). A fragments stay in LDS; an opaque
// asm redefinition of the LDS offset per tile iteration blocks LICM so the
// compiler cannot hoist (and then spill) them.
__global__ __launch_bounds__(256) void k_logits(
    const unsigned short* __restrict__ Ebf, const unsigned short* __restrict__ embbf,
    const float* __restrict__ out_b, const int* __restrict__ inputs,
    float* __restrict__ Pm, float* __restrict__ Ps, float* __restrict__ Pbv,
    int* __restrict__ Pbi, float* __restrict__ Ptv)
{
  __shared__ unsigned int lds_a[ROWT * 512];     // 32 rows x 1024 bf16 = 64 KB
  __shared__ float shm_m[8][ROWT], shm_s[8][ROWT], shm_bv[8][ROWT], shm_tv[8][ROWT];
  __shared__ int   shm_bi[8][ROWT];

  const int r0 = blockIdx.x * ROWT;              // step-row tile base
  const int c  = blockIdx.y;                     // vocab chunk
  const int c0 = c * CHUNK;

  { // stage E row-tile into LDS (128-bit loads): ROWT*128 = 4096 uint4
    const uint4* src = (const uint4*)(Ebf + (size_t)r0 * HID);
    uint4* dst = (uint4*)lds_a;
    for (int i = threadIdx.x; i < ROWT * 128; i += 256) dst[i] = src[i];
  }
  __syncthreads();

  const int wid  = threadIdx.x >> 5;
  const int lane = threadIdx.x & 31;
  const int half = lane >> 4;
  const int nl   = lane & 15;

  float rm[2][8], rs[2][8], rbv[2][8], rtv[2][8];
  int   rbi[2][8], tg[2][8];
  #pragma unroll
  for (int s = 0; s < 2; ++s) {
    #pragma unroll
    for (int v = 0; v < 8; ++v) {
      rm[s][v] = -__builtin_inff(); rs[s][v] = 0.f;
      rbv[s][v] = -__builtin_inff(); rbi[s][v] = 0x7fffffff;
      rtv[s][v] = -__builtin_inff();
      int r = r0 + s * 16 + v + 8 * half;                // C/D row for acc[s][v]
      tg[s][v] = (r < TSTEPS) ? inputs[r + 1] : -2;      // targets[t] = inputs[t+1]
    }
  }

  for (int wt = wid; wt < COLT; wt += 8) {
    const int n = c0 + wt * 16 + nl;                     // vocab column for this lane
    const uint4* bq = (const uint4*)(embbf + (size_t)n * HID);
    v8f acc[2];
    acc[0] = (v8f){0.f,0.f,0.f,0.f,0.f,0.f,0.f,0.f};
    acc[1] = (v8f){0.f,0.f,0.f,0.f,0.f,0.f,0.f,0.f};

    // Opaque per-iteration LDS base: blocks hoisting of A-frag ds loads.
    unsigned aoff = nl * 512;
    asm volatile("" : "+v"(aoff));
    const unsigned int* ar0 = lds_a + aoff;              // tile 0: rows r0..r0+15
    const unsigned int* ar1 = lds_a + aoff + 16 * 512;   // tile 1: rows r0+16..r0+31

    // B 32x16 bf16: lane n, K = kk + 16*half + 2j -> uint4 idx kk/8 + 2*half.
    // Pipeline depth 2: fragments for kk and kk+32 always in flight.
    Frag fb0, fb1;
    { const uint4* bp = bq + (half << 1);     fb0.q[0] = bp[0]; fb0.q[1] = bp[1]; }
    { const uint4* bp = bq + 4 + (half << 1); fb1.q[0] = bp[0]; fb1.q[1] = bp[1]; }

    #pragma unroll
    for (int kk = 0; kk < HID; kk += 32) {
      Frag fbn;
      if (kk + 64 < HID) {                               // prefetch B for kk+64
        const uint4* bp = bq + ((kk + 64) >> 3) + (half << 1);
        fbn.q[0] = bp[0];
        fbn.q[1] = bp[1];
      } else {
        fbn = fb1;                                       // dead in final iterations
      }
      // A 16x32 bf16 frags from LDS (both row tiles)
      Frag fa0, fa1;
      const uint4* a0 = (const uint4*)(ar0 + (kk >> 1) + (half << 3));
      fa0.q[0] = a0[0];
      fa0.q[1] = a0[1];
      const uint4* a1 = (const uint4*)(ar1 + (kk >> 1) + (half << 3));
      fa1.q[0] = a1[0];
      fa1.q[1] = a1[1];

      acc[0] = __builtin_amdgcn_wmma_f32_16x16x32_bf16(
          false, fa0.v, false, fb0.v, (short)0, acc[0], false, false);
      acc[1] = __builtin_amdgcn_wmma_f32_16x16x32_bf16(
          false, fa1.v, false, fb0.v, (short)0, acc[1], false, false);
      fb0 = fb1;
      fb1 = fbn;
    }

    const float ob = out_b[n];
    #pragma unroll
    for (int s = 0; s < 2; ++s) {
      #pragma unroll
      for (int v = 0; v < 8; ++v) {
        float tv = acc[s][v] + ob;                       // logit (row r0+16s+v+8h, col n)
        float mx = tv; int mi = n;
        float tt = (n == tg[s][v]) ? tv : -__builtin_inff();
        #pragma unroll
        for (int msk = 1; msk < 16; msk <<= 1) {         // butterfly over 16-lane half
          float ov = __shfl_xor(mx, msk, 32);
          int   oi = __shfl_xor(mi, msk, 32);
          if (ov > mx || (ov == mx && oi < mi)) { mx = ov; mi = oi; }
          tt = fmaxf(tt, __shfl_xor(tt, msk, 32));
        }
        float se = __expf(tv - mx);
        #pragma unroll
        for (int msk = 1; msk < 16; msk <<= 1) se += __shfl_xor(se, msk, 32);
        // online merge into running per-row state
        if (mx > rbv[s][v] || (mx == rbv[s][v] && mi < rbi[s][v])) { rbv[s][v] = mx; rbi[s][v] = mi; }
        float nm = fmaxf(rm[s][v], mx);
        rs[s][v] = rs[s][v] * __expf(rm[s][v] - nm) + se * __expf(mx - nm);
        rm[s][v] = nm;
        rtv[s][v] = fmaxf(rtv[s][v], tt);
      }
    }
  }

  if (nl == 0) {
    #pragma unroll
    for (int s = 0; s < 2; ++s) {
      #pragma unroll
      for (int v = 0; v < 8; ++v) {
        int m = s * 16 + v + 8 * half;
        shm_m[wid][m]  = rm[s][v];  shm_s[wid][m]  = rs[s][v];
        shm_bv[wid][m] = rbv[s][v]; shm_bi[wid][m] = rbi[s][v]; shm_tv[wid][m] = rtv[s][v];
      }
    }
  }
  __syncthreads();
  if (threadIdx.x < ROWT) {
    const int m = threadIdx.x;
    float M = -__builtin_inff(), S = 0.f, BV = -__builtin_inff(), TV = -__builtin_inff();
    int BI = 0x7fffffff;
    for (int w8 = 0; w8 < 8; ++w8) {
      float wm = shm_m[w8][m], ws2 = shm_s[w8][m], wbv = shm_bv[w8][m], wtv = shm_tv[w8][m];
      int wbi = shm_bi[w8][m];
      if (wbv > BV || (wbv == BV && wbi < BI)) { BV = wbv; BI = wbi; }
      float nm = fmaxf(M, wm);
      S = S * __expf(M - nm) + ws2 * __expf(wm - nm);
      M = nm;
      TV = fmaxf(TV, wtv);
    }
    const int r = r0 + m;
    if (r < TSTEPS) {
      const int pi = r * NCHUNK + c;
      Pm[pi] = M; Ps[pi] = S; Pbv[pi] = BV; Pbi[pi] = BI; Ptv[pi] = TV;
    }
  }
}

// ---------------- merge chunk partials per row ------------------------------
__global__ __launch_bounds__(256) void k_rowmerge(
    const float* __restrict__ Pm, const float* __restrict__ Ps,
    const float* __restrict__ Pbv, const int* __restrict__ Pbi,
    const float* __restrict__ Ptv, float* __restrict__ losses,
    float* __restrict__ out)
{
  const int t = blockIdx.x * 256 + threadIdx.x;
  if (t == 0) out[1] = 0.f;                    // outputs[0,0] stays 0
  if (t >= TSTEPS) return;
  float M = -__builtin_inff(), S = 0.f, BV = -__builtin_inff(), TV = -__builtin_inff();
  int BI = 0x7fffffff;
  for (int c2 = 0; c2 < NCHUNK; ++c2) {
    const int pi = t * NCHUNK + c2;
    float wm = Pm[pi], ws2 = Ps[pi], wbv = Pbv[pi], wtv = Ptv[pi];
    int wbi = Pbi[pi];
    if (wbv > BV || (wbv == BV && wbi < BI)) { BV = wbv; BI = wbi; }
    float nm = fmaxf(M, wm);
    S = S * __expf(M - nm) + ws2 * __expf(wm - nm);
    M = nm;
    TV = fmaxf(TV, wtv);
  }
  losses[t] = (M + logf(S)) - TV;              // logsumexp - target_logit
  out[2 + t] = (float)BI;                      // outputs[0, 1+t] = argmax
}

// ---------------- sum losses -> out[0] --------------------------------------
__global__ __launch_bounds__(256) void k_sumloss(const float* __restrict__ losses,
                                                 float* __restrict__ out) {
  __shared__ float red[256];
  float a = 0.f;
  for (int i = threadIdx.x; i < TSTEPS; i += 256) a += losses[i];
  red[threadIdx.x] = a;
  __syncthreads();
  for (int s = 128; s > 0; s >>= 1) {
    if (threadIdx.x < s) red[threadIdx.x] += red[threadIdx.x + s];
    __syncthreads();
  }
  if (threadIdx.x == 0) out[0] = red[0];
}

extern "C" void kernel_launch(void* const* d_in, const int* in_sizes, int n_in,
                              void* d_out, int out_size, void* d_ws, size_t ws_size,
                              hipStream_t stream) {
  (void)in_sizes; (void)n_in; (void)out_size; (void)ws_size;
  const int*   inputs = (const int*)  d_in[0];
  const float* z      = (const float*)d_in[1];
  const float* emb_W  = (const float*)d_in[2];
  const float* out_b  = (const float*)d_in[3];
  const float* h2e_W  = (const float*)d_in[4];
  const float* h2e_b  = (const float*)d_in[5];
  const float* W_ih   = (const float*)d_in[6];
  const float* W_hh   = (const float*)d_in[7];
  const float* b_ih   = (const float*)d_in[8];
  const float* b_hh   = (const float*)d_in[9];
  float* out = (float*)d_out;

  char* w = (char*)d_ws;
  size_t off = 0;
  auto take = [&](size_t bytes) {
    char* p = w + off;
    off = (off + bytes + 255) & ~(size_t)255;
    return p;
  };
  unsigned short* emb_bf = (unsigned short*)take((size_t)VOCAB * HID * 2);
  unsigned short* wih_bf = (unsigned short*)take((size_t)3 * HID * HID * 2);
  unsigned short* whh_bf = (unsigned short*)take((size_t)3 * HID * HID * 2);
  unsigned short* h2e_bf = (unsigned short*)take((size_t)HID * HID * 2);
  unsigned short* Ebf    = (unsigned short*)take((size_t)1024 * HID * 2);
  float* gA     = (float*)take(2 * 3072 * 4);
  float* gB     = (float*)take(2 * 3072 * 4);
  float* Pm     = (float*)take((size_t)1024 * NCHUNK * 4);
  float* Ps     = (float*)take((size_t)1024 * NCHUNK * 4);
  float* Pbv    = (float*)take((size_t)1024 * NCHUNK * 4);
  int*   Pbi    = (int*)  take((size_t)1024 * NCHUNK * 4);
  float* Ptv    = (float*)take((size_t)1024 * NCHUNK * 4);
  float* losses = (float*)take(1024 * 4);
  int*   bar    = (int*)  take(256);

  k_init<<<1, 64, 0, stream>>>(bar);
  k_f32bf16<<<2048, 256, 0, stream>>>(emb_W, emb_bf, (size_t)VOCAB * HID);
  k_f32bf16<<<512, 256, 0, stream>>>(W_ih, wih_bf, (size_t)3 * HID * HID);
  k_f32bf16<<<512, 256, 0, stream>>>(W_hh, whh_bf, (size_t)3 * HID * HID);
  k_f32bf16<<<256, 256, 0, stream>>>(h2e_W, h2e_bf, (size_t)HID * HID);

  k_gru<<<NB1, 256, 0, stream>>>(inputs, z, emb_W, h2e_b, b_ih, b_hh,
                                 wih_bf, whh_bf, h2e_bf, Ebf, gA, gB, bar);

  dim3 g2(1024 / ROWT, NCHUNK);
  k_logits<<<g2, 256, 0, stream>>>(Ebf, emb_bf, out_b, inputs, Pm, Ps, Pbv, Pbi, Ptv);
  k_rowmerge<<<4, 256, 0, stream>>>(Pm, Ps, Pbv, Pbi, Ptv, losses, out);
  k_sumloss<<<1, 256, 0, stream>>>(losses, out);
}
